// RegionProposeNetwork_71124658421969
// MI455X (gfx1250) — matrix operand order repaired and checked by reading, not compiled, not run on hardware
//
#include <hip/hip_runtime.h>
#include <hip/hip_bf16.h>

// ---------------------------------------------------------------------------
// RPN head + loss for MI455X (gfx1250): bf16 WMMA implicit-GEMM conv pipeline
// with async global->LDS weight staging (ASYNCcnt path).
// ---------------------------------------------------------------------------

#define HW       62
#define MB       3844      // 62*62 spatial positions per batch
#define MTILES   31        // ceil(3844/128)
#define MPAD     3968      // 31*128 padded rows per batch
#define CIN      256
#define COUT     256
#define ANUM     9
#define NANCH    34596     // 3844*9
#define BATCH    16
#define NGT      8

typedef __attribute__((ext_vector_type(16))) __bf16 v16bf;
typedef __attribute__((ext_vector_type(8)))  __bf16 v8bf;
typedef __attribute__((ext_vector_type(8)))  float  v8f;

// ---- workspace layout (bytes, 256-aligned) --------------------------------
#define OFF_WPACK  ((size_t)0)            // 9*256*256 bf16            = 1,179,648
#define OFF_W2     ((size_t)1179648)      // 48*256 bf16               =    24,576
#define OFF_XPACK  ((size_t)1204224)      // 16*3968*256 bf16          = 32,505,856
#define OFF_CLS    ((size_t)33710080)     // 16*34596 f32              =  2,214,144
#define OFF_BOX    ((size_t)35924224)     // 16*34596*4 f32            =  8,856,576
#define OFF_MAX    ((size_t)44780800)     // 16*8 u32 (+pad)           =       512
#define OFF_ACC    ((size_t)44781312)     // 8 f32 accumulators

// ---- d_out layout (float elements) ----------------------------------------
#define O_PROP   ((size_t)2)
#define O_MASK   ((size_t)2214146)
#define O_GT     ((size_t)6641858)

__device__ __forceinline__ unsigned pack2bf(float a, float b) {
    unsigned short ua = __builtin_bit_cast(unsigned short, (__bf16)a);
    unsigned short ub = __builtin_bit_cast(unsigned short, (__bf16)b);
    return (unsigned)ua | ((unsigned)ub << 16);
}

// ===========================================================================
// Kernel 0: zero per-launch scratch state (graph-replay safe)
// ===========================================================================
__global__ void init_scratch(unsigned* maxbits, float* accum) {
    int t = threadIdx.x;
    if (t < 128) maxbits[t] = 0u;
    if (t < 8)   accum[t]   = 0.0f;
}

// ===========================================================================
// Kernel 1a: conv_w (OIHW fp32) -> Wpack[kidx][co][ci] bf16
// ===========================================================================
__global__ __launch_bounds__(256) void pack_w(const float* __restrict__ conv_w,
                                              __bf16* __restrict__ Wpack) {
    int idx = blockIdx.x * 256 + threadIdx.x;           // [kidx][co][ci]
    if (idx >= 9 * 256 * 256) return;
    int ci   = idx & 255;
    int co   = (idx >> 8) & 255;
    int kidx = idx >> 16;
    int kh = kidx / 3, kw = kidx % 3;
    float v = conv_w[((co * 256 + ci) * 3 + kh) * 3 + kw];
    Wpack[idx] = (__bf16)v;
}

// Kernel 1b: cls_w(9,256) + box_w(36,256) -> W2[48][256] bf16 (rows 45..47 = 0)
__global__ __launch_bounds__(256) void pack_w2(const float* __restrict__ cls_w,
                                               const float* __restrict__ box_w,
                                               __bf16* __restrict__ W2) {
    int idx = blockIdx.x * 256 + threadIdx.x;
    if (idx >= 48 * 256) return;
    int k = idx & 255, n = idx >> 8;
    float v = 0.0f;
    if (n < 9)       v = cls_w[n * 256 + k];
    else if (n < 45) v = box_w[(n - 9) * 256 + k];
    W2[idx] = (__bf16)v;
}

// ===========================================================================
// Kernel 2: 3x3 VALID conv as implicit GEMM via v_wmma_f32_16x16x32_bf16.
// Workgroup = 8 waves, tile = 128 spatial (M) x 128 Cout (N), K = 2304.
// Wave tile = 32M x 64N -> 8 WMMAs per K=32 step (acc[2][4] v8f).
// B tile staged via GLOBAL_LOAD_ASYNC_TO_LDS_B128 (ASYNCcnt), A staged with
// fp32->bf16 packed b32 DS stores.
// xpack output layout: [b][m(0..3967)][co] bf16 (rows >= 3844 are zero).
// ===========================================================================
__global__ __launch_bounds__(256)
void conv3x3_wmma(const float* __restrict__ feat, const float* __restrict__ convb,
                  const __bf16* __restrict__ Wpack, __bf16* __restrict__ xpack) {
    __shared__ __bf16 As[128 * 40];   // 128 rows x 32 k, stride 40 (bank pad)
    __shared__ __bf16 Bs[128 * 40];   // 128 cols x 32 k, stride 40

    const int b     = blockIdx.y;
    const int mtile = blockIdx.x % MTILES;
    const int ntile = blockIdx.x / MTILES;
    const int tid   = threadIdx.x;
    const int lane  = tid & 31;
    const int wave  = tid >> 5;
    const int wm    = wave & 3;       // 4 M sub-tiles of 32
    const int wn    = wave >> 2;      // 2 N sub-tiles of 64

    v8f acc[2][4];
#pragma unroll
    for (int i = 0; i < 2; ++i)
#pragma unroll
        for (int j = 0; j < 4; ++j) { v8f z = {}; acc[i][j] = z; }

    // A-fill mapping: thread -> (m_local 0..127, ci group of 16)
    const int a_m  = tid & 127;
    const int a_cg = tid >> 7;        // 0..1
    const int m    = mtile * 128 + a_m;
    const bool mvalid = (m < MB);
    const int oh = m / HW, ow = m % HW;

    // B-fill mapping: thread -> (n_local 0..127, 16-element half)
    const int b_n = tid & 127;
    const int b_h = tid >> 7;         // 0..1
    const int co_fill = ntile * 128 + b_n;
    const unsigned lds_b =
        (unsigned)(unsigned long long)(&Bs[b_n * 40 + b_h * 16]);

    const int fr_row = wm * 32 + (lane & 15);
    const int fr_kb  = (lane >> 4) * 16;   // A fragment k-base (elements)
    const int fr_kb2 = (lane >> 4) * 16;   // B fragment k-base (elements)

    for (int kidx = 0; kidx < 9; ++kidx) {
        const int kh = kidx / 3, kw = kidx % 3;
        const float* fbase = feat + ((size_t)b * CIN) * 4096 + (oh + kh) * 64 + (ow + kw);
        const __bf16* wbase = Wpack + ((size_t)(kidx * COUT + co_fill)) * CIN;

        for (int cc = 0; cc < 8; ++cc) {
            const int ci0 = cc * 32;

            // ---- B tile: async DMA straight into LDS (32 B per thread) ----
            {
                unsigned long long gaddr =
                    (unsigned long long)(wbase + ci0 + b_h * 16);
                asm volatile(
                    "global_load_async_to_lds_b128 %0, %1, off\n\t"
                    "global_load_async_to_lds_b128 %0, %1, off offset:16"
                    :: "v"(lds_b), "v"(gaddr) : "memory");
            }

            // ---- A tile: im2col gather, fp32 -> bf16, packed b32 stores ----
#pragma unroll
            for (int q = 0; q < 16; q += 2) {
                const int ci_off = a_cg * 16 + q;
                float v0 = mvalid ? fbase[(size_t)(ci0 + ci_off) * 4096] : 0.0f;
                float v1 = mvalid ? fbase[(size_t)(ci0 + ci_off + 1) * 4096] : 0.0f;
                *(unsigned*)&As[a_m * 40 + ci_off] = pack2bf(v0, v1);
            }
            // hint next K-chunk into cache (global_prefetch_b8)
            if (mvalid && cc < 7)
                __builtin_prefetch(fbase + (size_t)(ci0 + 32 + a_cg * 16) * 4096, 0, 1);

            asm volatile("s_wait_asynccnt 0" ::: "memory");
            __syncthreads();

            // ---- fragments + 8 WMMAs ----
            v8bf a0lo = *(const v8bf*)&As[fr_row * 40 + fr_kb];
            v8bf a0hi = *(const v8bf*)&As[fr_row * 40 + fr_kb + 8];
            v8bf a1lo = *(const v8bf*)&As[(fr_row + 16) * 40 + fr_kb];
            v8bf a1hi = *(const v8bf*)&As[(fr_row + 16) * 40 + fr_kb + 8];
            v16bf afrag0 = __builtin_shufflevector(a0lo, a0hi,
                0, 1, 2, 3, 4, 5, 6, 7, 8, 9, 10, 11, 12, 13, 14, 15);
            v16bf afrag1 = __builtin_shufflevector(a1lo, a1hi,
                0, 1, 2, 3, 4, 5, 6, 7, 8, 9, 10, 11, 12, 13, 14, 15);
#pragma unroll
            for (int nn = 0; nn < 4; ++nn) {
                const int col = wn * 64 + nn * 16 + (lane & 15);
                v16bf bfrag = *(const v16bf*)&Bs[col * 40 + fr_kb2];
                acc[0][nn] = __builtin_amdgcn_wmma_f32_16x16x32_bf16(
                    false, afrag0, false, bfrag, (short)0, acc[0][nn], false, false);
                acc[1][nn] = __builtin_amdgcn_wmma_f32_16x16x32_bf16(
                    false, afrag1, false, bfrag, (short)0, acc[1][nn], false, false);
            }
            __syncthreads();
        }
    }

    // ---- epilogue: +bias, cvt bf16, store [b][m][co] ----
#pragma unroll
    for (int nn = 0; nn < 4; ++nn) {
        const int co = ntile * 128 + wn * 64 + nn * 16 + (lane & 15);
        const float bias = convb[co];
#pragma unroll
        for (int mi = 0; mi < 2; ++mi) {
#pragma unroll
            for (int r = 0; r < 8; ++r) {
                const int mrow = mtile * 128 + wm * 32 + mi * 16 + r + 8 * (lane >> 4);
                const float v = (mrow < MB) ? (acc[mi][nn][r] + bias) : 0.0f;
                xpack[((size_t)b * MPAD + mrow) * COUT + co] = (__bf16)v;
            }
        }
    }
}

// ===========================================================================
// Kernel 3: fused cls(9)+box(36) 1x1 convs: GEMM [16*3968 x 48 x 256] via WMMA.
// 8 waves/block, 16 rows/wave, weights resident in LDS.
// ===========================================================================
__global__ __launch_bounds__(256)
void conv1x1_wmma(const __bf16* __restrict__ xpack, const __bf16* __restrict__ W2,
                  const float* __restrict__ clsb, const float* __restrict__ boxb,
                  float* __restrict__ cls_logit, float* __restrict__ pred_off) {
    __shared__ __bf16 Ws[48 * 264];   // 48 rows x 256 k, stride 264 (pad)

    const int tid  = threadIdx.x;
    const int lane = tid & 31;
    const int wave = tid >> 5;

    // stage all 48x256 weights into LDS (768 chunks of 16 bf16)
    for (int c = tid; c < 768; c += 256) {
        const int n = c >> 4, h = c & 15;
        v16bf wv = *(const v16bf*)&W2[n * 256 + h * 16];
        *(v16bf*)&Ws[n * 264 + h * 16] = wv;
    }
    __syncthreads();

    const size_t row_base = (size_t)blockIdx.x * 128 + (size_t)wave * 16;
    const size_t rowa = row_base + (lane & 15);

    v8f acc[3];
#pragma unroll
    for (int i = 0; i < 3; ++i) { v8f z = {}; acc[i] = z; }

#pragma unroll
    for (int ks = 0; ks < 8; ++ks) {
        const int kb = ks * 32 + (lane >> 4) * 16;
        v8bf alo = *(const v8bf*)&xpack[rowa * 256 + kb];
        v8bf ahi = *(const v8bf*)&xpack[rowa * 256 + kb + 8];
        v16bf afrag = __builtin_shufflevector(alo, ahi,
            0, 1, 2, 3, 4, 5, 6, 7, 8, 9, 10, 11, 12, 13, 14, 15);
#pragma unroll
        for (int nn = 0; nn < 3; ++nn) {
            const int col = nn * 16 + (lane & 15);
            v16bf bfrag = *(const v16bf*)&Ws[col * 264 + ks * 32 + (lane >> 4) * 16];
            acc[nn] = __builtin_amdgcn_wmma_f32_16x16x32_bf16(
                false, afrag, false, bfrag, (short)0, acc[nn], false, false);
        }
    }

    // epilogue: scatter into reference (h,w,A) / (anchor,4) flattening
#pragma unroll
    for (int nn = 0; nn < 3; ++nn) {
        const int n = nn * 16 + (lane & 15);
#pragma unroll
        for (int r = 0; r < 8; ++r) {
            const size_t mg = row_base + r + 8 * (lane >> 4);
            const int bb = (int)(mg / MPAD);
            const int m  = (int)(mg % MPAD);
            if (m >= MB) continue;
            float v = acc[nn][r];
            if (n < 9) {
                v += clsb[n];
                cls_logit[(size_t)bb * NANCH + (size_t)m * 9 + n] = v;
            } else if (n < 45) {
                const int c = n - 9;                 // channel = 4*a + j
                v += boxb[c];
                const size_t anch = (size_t)m * 9 + (c >> 2);
                pred_off[((size_t)bb * NANCH + anch) * 4 + (c & 3)] = v;
            }
        }
    }
}

// ===========================================================================
// Anchor / IoU helpers (identical arithmetic in both passes -> bit-equal iou)
// ===========================================================================
__device__ __forceinline__ void anchor_box(int anch, float ax[4]) {
    const int a  = anch % 9;
    const int mm = anch / 9;
    const int i1 = mm % HW;   // second grid axis (ctr-Y)
    const int i0 = mm / HW;   // first grid axis  (ctr-X)
    const float sc[3] = {2.0f, 4.0f, 6.0f};
    const float rt[3] = {0.5f, 1.0f, 2.0f};
    const float s = sc[a % 3];
    const float r = rt[a / 3];
    const float w = s * r, h = s;
    const float cx = i0 + 0.5f, cy = i1 + 0.5f;
    ax[0] = fminf(fmaxf(cx - 0.5f * w, 0.0f), (float)HW);
    ax[1] = fminf(fmaxf(cy - 0.5f * h, 0.0f), (float)HW);
    ax[2] = fminf(fmaxf(cx + 0.5f * w, 0.0f), (float)HW);
    ax[3] = fminf(fmaxf(cy + 0.5f * h, 0.0f), (float)HW);
}

__device__ __forceinline__ void load_gt(const float* __restrict__ gt_boxes, int b,
                                        float (&g)[NGT][4]) {
#pragma unroll
    for (int j = 0; j < NGT; ++j)
#pragma unroll
        for (int c = 0; c < 4; ++c)
            g[j][c] = gt_boxes[((size_t)(b * NGT + j)) * 4 + c] * 0.125f;  // /8 (512//62)
}

__device__ __forceinline__ float iou_one(const float ax[4], const float g[4]) {
    const float ix1 = fmaxf(ax[0], g[0]);
    const float iy1 = fmaxf(ax[1], g[1]);
    const float ix2 = fminf(ax[2], g[2]);
    const float iy2 = fminf(ax[3], g[3]);
    const float inter  = fmaxf(ix2 - ix1, 0.0f) * fmaxf(iy2 - iy1, 0.0f);
    const float area_a = (ax[2] - ax[0]) * (ax[3] - ax[1]);
    const float area_g = (g[2] - g[0]) * (g[3] - g[1]);
    const float uni = area_a + area_g - inter;
    return inter / fmaxf(uni, 1e-9f);
}

__device__ __forceinline__ float softplusf(float x) {
    return fmaxf(x, 0.0f) + log1pf(expf(-fabsf(x)));
}

// ===========================================================================
// Kernel 4a: per-(b, gt) max IoU over all anchors (block reduce + atomicMax)
// ===========================================================================
__global__ __launch_bounds__(256)
void iou_max_kernel(const float* __restrict__ gt_boxes, unsigned* __restrict__ maxbits) {
    __shared__ float sm[256];
    const int b    = blockIdx.y;
    const int anch = blockIdx.x * 256 + threadIdx.x;
    const int tid  = threadIdx.x;
    const bool valid = (anch < NANCH);

    float g[NGT][4];
    load_gt(gt_boxes, b, g);
    float ax[4] = {0.f, 0.f, 0.f, 0.f};
    if (valid) anchor_box(anch, ax);

    float iou[NGT];
#pragma unroll
    for (int j = 0; j < NGT; ++j) iou[j] = valid ? iou_one(ax, g[j]) : 0.0f;

#pragma unroll
    for (int j = 0; j < NGT; ++j) {
        sm[tid] = iou[j];
        __syncthreads();
        for (int s = 128; s > 0; s >>= 1) {
            if (tid < s) sm[tid] = fmaxf(sm[tid], sm[tid + s]);
            __syncthreads();
        }
        if (tid == 0) atomicMax(&maxbits[b * NGT + j], __float_as_uint(sm[0]));
        __syncthreads();
    }
}

// ===========================================================================
// Kernel 4b: assignments, proposals, mask/gt outputs, loss partial sums
// ===========================================================================
__global__ __launch_bounds__(256)
void assign_loss_kernel(const float* __restrict__ gt_boxes, const int* __restrict__ gt_class,
                        const float* __restrict__ cls_logit, const float* __restrict__ pred_off,
                        const unsigned* __restrict__ maxbits, float* __restrict__ accum,
                        float* __restrict__ out_prop, float* __restrict__ out_mask,
                        float* __restrict__ out_gt) {
    __shared__ float sm[256];
    const int b    = blockIdx.y;
    const int anch = blockIdx.x * 256 + threadIdx.x;
    const int tid  = threadIdx.x;
    const bool valid = (anch < NANCH);

    float g[NGT][4];
    load_gt(gt_boxes, b, g);
    float ax[4] = {0.f, 0.f, 1.f, 1.f};
    if (valid) anchor_box(anch, ax);

    const float acx = 0.5f * (ax[0] + ax[2]);
    const float acy = 0.5f * (ax[1] + ax[3]);
    const float aw  = ax[2] - ax[0];
    const float ah  = ax[3] - ax[1];
    const float law = logf(aw), lah = logf(ah);

    float cls = 0.0f, po[4] = {0.f, 0.f, 0.f, 0.f};
    if (valid) {
        cls = cls_logit[(size_t)b * NANCH + anch];
#pragma unroll
        for (int c = 0; c < 4; ++c)
            po[c] = pred_off[((size_t)b * NANCH + anch) * 4 + c];
    }
    const float sp_neg = softplusf(-cls);
    const float sp_pos = softplusf(cls);

    float pos_cnt = 0.f, neg_cnt = 0.f, reg_sum = 0.f, pos_loss = 0.f, neg_loss = 0.f;
    float best = -1.0f; int bestj = 0;

#pragma unroll
    for (int j = 0; j < NGT; ++j) {
        const float iou = valid ? iou_one(ax, g[j]) : 0.0f;
        if (valid) {
            const float mx = __uint_as_float(maxbits[b * NGT + j]);
            const bool pos = ((iou == mx) && (mx > 0.0f)) || (iou > 0.7f);
            const bool neg = (iou < 0.3f);
            out_mask[((size_t)b * NANCH + anch) * NGT + j] = pos ? 1.0f : 0.0f;
            if (pos) {
                pos_cnt  += 1.0f;
                pos_loss += sp_neg;
                const float gcx = 0.5f * (g[j][0] + g[j][2]);
                const float gcy = 0.5f * (g[j][1] + g[j][3]);
                const float gw  = g[j][2] - g[j][0];
                const float gh  = g[j][3] - g[j][1];
                const float t[4] = {(gcx - acx) / aw, (gcy - acy) / ah,
                                    logf(gw) - law, logf(gh) - lah};
#pragma unroll
                for (int c = 0; c < 4; ++c) {
                    const float d  = po[c] - t[c];
                    const float ad = fabsf(d);
                    reg_sum += (ad < 1.0f) ? 0.5f * d * d : ad - 0.5f;
                }
            }
            if (neg) { neg_cnt += 1.0f; neg_loss += sp_pos; }
            if (iou > best) { best = iou; bestj = j; }
        }
    }

    if (valid) {
        out_gt[(size_t)b * NANCH + anch] = (float)gt_class[b * NGT + bestj];
        const float pcx = acx + po[0] * aw;
        const float pcy = acy + po[1] * ah;
        const float pw  = aw * expf(po[2]);
        const float ph  = ah * expf(po[3]);
        const float ps  = 64.0f / 62.0f;
        float* pp = out_prop + ((size_t)b * NANCH + anch) * 4;
        pp[0] = (pcx - 0.5f * pw) * ps;
        pp[1] = (pcy - 0.5f * ph) * ps;
        pp[2] = (pcx + 0.5f * pw) * ps;
        pp[3] = (pcy + 0.5f * ph) * ps;
    }

    // block-reduce the five partial sums, then one atomicAdd each
    float sums[5] = {pos_cnt, neg_cnt, reg_sum, pos_loss, neg_loss};
#pragma unroll
    for (int k = 0; k < 5; ++k) {
        sm[tid] = sums[k];
        __syncthreads();
        for (int s = 128; s > 0; s >>= 1) {
            if (tid < s) sm[tid] += sm[tid + s];
            __syncthreads();
        }
        if (tid == 0) atomicAdd(&accum[k], sm[0]);
        __syncthreads();
    }
}

// ===========================================================================
// Kernel 5: scalar loss finalization
// ===========================================================================
__global__ void finalize_kernel(const float* __restrict__ accum, float* __restrict__ out) {
    const float np = fmaxf(accum[0], 1.0f);
    const float nn = fmaxf(accum[1], 1.0f);
    out[0] = 0.5f * (accum[3] / np + accum[4] / nn);   // cls_loss
    out[1] = accum[2] / (np * 4.0f);                   // reg_loss
}

// ===========================================================================
extern "C" void kernel_launch(void* const* d_in, const int* in_sizes, int n_in,
                              void* d_out, int out_size, void* d_ws, size_t ws_size,
                              hipStream_t stream) {
    const float* feat     = (const float*)d_in[0];
    const float* gt_boxes = (const float*)d_in[1];
    const int*   gt_class = (const int*)  d_in[2];
    const float* conv_w   = (const float*)d_in[3];
    const float* conv_b   = (const float*)d_in[4];
    const float* cls_w    = (const float*)d_in[5];
    const float* cls_b    = (const float*)d_in[6];
    const float* box_w    = (const float*)d_in[7];
    const float* box_b    = (const float*)d_in[8];
    (void)in_sizes; (void)n_in; (void)out_size; (void)ws_size;

    char* ws = (char*)d_ws;
    __bf16*   Wpack     = (__bf16*)(ws + OFF_WPACK);
    __bf16*   W2        = (__bf16*)(ws + OFF_W2);
    __bf16*   xpack     = (__bf16*)(ws + OFF_XPACK);
    float*    cls_logit = (float*)(ws + OFF_CLS);
    float*    pred_off  = (float*)(ws + OFF_BOX);
    unsigned* maxbits   = (unsigned*)(ws + OFF_MAX);
    float*    accum     = (float*)(ws + OFF_ACC);
    float*    out       = (float*)d_out;

    init_scratch<<<1, 128, 0, stream>>>(maxbits, accum);
    pack_w <<<2304, 256, 0, stream>>>(conv_w, Wpack);
    pack_w2<<<48,   256, 0, stream>>>(cls_w, box_w, W2);

    conv3x3_wmma<<<dim3(MTILES * 2, BATCH), 256, 0, stream>>>(feat, conv_b, Wpack, xpack);
    conv1x1_wmma<<<496, 256, 0, stream>>>(xpack, W2, cls_b, box_b, cls_logit, pred_off);

    const int ablocks = (NANCH + 255) / 256;  // 136
    iou_max_kernel    <<<dim3(ablocks, BATCH), 256, 0, stream>>>(gt_boxes, maxbits);
    assign_loss_kernel<<<dim3(ablocks, BATCH), 256, 0, stream>>>(
        gt_boxes, gt_class, cls_logit, pred_off, maxbits, accum,
        out + O_PROP, out + O_MASK, out + O_GT);
    finalize_kernel<<<1, 1, 0, stream>>>(accum, out);
}
